// QExpRNNCell_4028679323712
// MI455X (gfx1250) — matrix operand order repaired and checked by reading, not compiled, not run on hardware
//
#include <hip/hip_runtime.h>
#include <hip/hip_bf16.h>
#include <math.h>

typedef __attribute__((ext_vector_type(16))) _Float16 v16h;
typedef __attribute__((ext_vector_type(8)))  _Float16 v8h;
typedef __attribute__((ext_vector_type(4)))  _Float16 v4h;
typedef __attribute__((ext_vector_type(8)))  float    v8f;

#define D_DIM 1024
#define TILE_M 128
#define TILE_N 128
#define TILE_K 32
#define LDSW 40   // halves per LDS row: 32 data + 8 pad (80B, 16B-aligned, bank-friendly)

// ---------------- scale init + abs-max reduction ----------------
__global__ __launch_bounds__(64) void k_zero_scales(unsigned* slots) {
    if (threadIdx.x < 2) slots[threadIdx.x] = 0u;
}

__global__ __launch_bounds__(256) void k_absmax(const float* __restrict__ w,
                                                unsigned n4, unsigned* __restrict__ slot) {
    __shared__ float red[256];
    const float4* p = (const float4*)w;
    float m = 0.0f;
    for (unsigned i = blockIdx.x * 256u + threadIdx.x; i < n4; i += gridDim.x * 256u) {
        float4 v = p[i];
        m = fmaxf(m, fmaxf(fmaxf(fabsf(v.x), fabsf(v.y)),
                           fmaxf(fabsf(v.z), fabsf(v.w))));
    }
    red[threadIdx.x] = m;
    __syncthreads();
    for (int s = 128; s > 0; s >>= 1) {
        if ((int)threadIdx.x < s) red[threadIdx.x] = fmaxf(red[threadIdx.x], red[threadIdx.x + s]);
        __syncthreads();
    }
    if (threadIdx.x == 0) atomicMax(slot, __float_as_uint(red[0]));  // |w| >= 0: bit compare OK
}

// ---------------- WMMA GEMM:  out[B,1024] = act[B,1024] x quant(W)[1024,1024]^T + bias ----
__global__ __launch_bounds__(256) void k_gemm_qw(const float* __restrict__ act,
                                                 const float* __restrict__ W,
                                                 const float* __restrict__ bias,
                                                 const unsigned* __restrict__ scaleSlot,
                                                 float* __restrict__ out) {
    __shared__ __align__(16) _Float16 Alds[TILE_M * LDSW];
    __shared__ __align__(16) _Float16 Blds[TILE_N * LDSW];

    const int tid     = threadIdx.x;
    const int rowBase = blockIdx.y * TILE_M;
    const int colBase = blockIdx.x * TILE_N;

    const float maxabs = __uint_as_float(*scaleSlot);
    const float scale  = maxabs * (1.0f / 7.0f);           // n = 2^(4-1)-1 = 7
    const float inv    = (maxabs > 0.0f) ? 7.0f / maxabs : 0.0f;

    const int wave    = tid >> 5;
    const int lane    = tid & 31;
    const int warpRow = (wave & 3) * 32;                   // 4 row-blocks of 32
    const int warpCol = (wave >> 2) * 64;                  // 2 col-blocks of 64
    const int mlane   = lane & 15;
    const int ksel    = lane >> 4;                         // 0: lanes 0-15, 1: lanes 16-31

    v8f acc[2][4];
    #pragma unroll
    for (int mb = 0; mb < 2; ++mb)
        #pragma unroll
        for (int nb = 0; nb < 4; ++nb)
            acc[mb][nb] = (v8f)(0.0f);

    const int ldRow = tid >> 3;        // 0..31, x4 passes -> 128 rows
    const int ldCol = (tid & 7) * 4;   // 0..28, float4 per thread

    for (int k0 = 0; k0 < D_DIM; k0 += TILE_K) {
        __syncthreads();
        #pragma unroll
        for (int p = 0; p < 4; ++p) {
            const int r = ldRow + p * 32;
            // activation tile -> f16
            float4 va = *(const float4*)(act + (size_t)(rowBase + r) * D_DIM + k0 + ldCol);
            v4h ha = { (_Float16)va.x, (_Float16)va.y, (_Float16)va.z, (_Float16)va.w };
            *(v4h*)(Alds + r * LDSW + ldCol) = ha;
            // weight tile -> fake-quant (round-to-nearest-even) -> f16 (exact: +/-k*scale)
            float4 vw = *(const float4*)(W + (size_t)(colBase + r) * D_DIM + k0 + ldCol);
            v4h hw = { (_Float16)(rintf(vw.x * inv) * scale),
                       (_Float16)(rintf(vw.y * inv) * scale),
                       (_Float16)(rintf(vw.z * inv) * scale),
                       (_Float16)(rintf(vw.w * inv) * scale) };
            *(v4h*)(Blds + r * LDSW + ldCol) = hw;
        }
        __syncthreads();

        // A fragments: lane holds row (mlane), K chunks [ksel*8 .. +7] and [16+ksel*8 .. +7]
        v16h afr[2];
        #pragma unroll
        for (int mb = 0; mb < 2; ++mb) {
            const _Float16* ap = Alds + (warpRow + mb * 16 + mlane) * LDSW + ksel * 8;
            v8h a0 = *(const v8h*)ap;
            v8h a1 = *(const v8h*)(ap + 16);
            afr[mb] = __builtin_shufflevector(a0, a1, 0,1,2,3,4,5,6,7,8,9,10,11,12,13,14,15);
        }
        // B fragments: lane holds col (mlane), K chunks [ksel*8 .. +7] and [16+ksel*8 .. +7]
        v16h bfr[4];
        #pragma unroll
        for (int nb = 0; nb < 4; ++nb) {
            const _Float16* bp = Blds + (warpCol + nb * 16 + mlane) * LDSW + ksel * 8;
            v8h b0 = *(const v8h*)bp;
            v8h b1 = *(const v8h*)(bp + 16);
            bfr[nb] = __builtin_shufflevector(b0, b1, 0,1,2,3,4,5,6,7,8,9,10,11,12,13,14,15);
        }
        #pragma unroll
        for (int mb = 0; mb < 2; ++mb)
            #pragma unroll
            for (int nb = 0; nb < 4; ++nb)
                acc[mb][nb] = __builtin_amdgcn_wmma_f32_16x16x32_f16(
                    false, afr[mb], false, bfr[nb], (short)0, acc[mb][nb], false, false);
    }

    // Epilogue: D layout — VGPR r: lanes 0-15 M=r, lanes 16-31 M=r+8; N = lane&15
    #pragma unroll
    for (int mb = 0; mb < 2; ++mb) {
        #pragma unroll
        for (int nb = 0; nb < 4; ++nb) {
            const int col = colBase + warpCol + nb * 16 + mlane;
            const float b = bias[col];
            #pragma unroll
            for (int r = 0; r < 8; ++r) {
                const int row = rowBase + warpRow + mb * 16 + ksel * 8 + r;
                out[(size_t)row * D_DIM + col] = acc[mb][nb][r] + b;
            }
        }
    }
}

// ---------------- fused LayerNorm(x) + LayerNorm(h) + tanh ----------------
__global__ __launch_bounds__(256) void k_ln_tanh(const float* __restrict__ X,
                                                 const float* __restrict__ H,
                                                 float* __restrict__ out) {
    __shared__ float sx[256], sxx[256], sh[256], shh[256];
    const int row = blockIdx.x;
    const int t   = threadIdx.x;
    float4 x = ((const float4*)(X + (size_t)row * D_DIM))[t];
    float4 h = ((const float4*)(H + (size_t)row * D_DIM))[t];
    sx[t]  = x.x + x.y + x.z + x.w;
    sxx[t] = x.x * x.x + x.y * x.y + x.z * x.z + x.w * x.w;
    sh[t]  = h.x + h.y + h.z + h.w;
    shh[t] = h.x * h.x + h.y * h.y + h.z * h.z + h.w * h.w;
    __syncthreads();
    for (int s = 128; s > 0; s >>= 1) {
        if (t < s) {
            sx[t]  += sx[t + s];  sxx[t] += sxx[t + s];
            sh[t]  += sh[t + s];  shh[t] += shh[t + s];
        }
        __syncthreads();
    }
    const float inv_n = 1.0f / (float)D_DIM;
    const float mux = sx[0] * inv_n;
    const float muh = sh[0] * inv_n;
    const float rsx = rsqrtf(fmaxf(sxx[0] * inv_n - mux * mux, 0.0f) + 1e-5f);
    const float rsh = rsqrtf(fmaxf(shh[0] * inv_n - muh * muh, 0.0f) + 1e-5f);
    float4 o;
    o.x = tanhf((x.x - mux) * rsx + (h.x - muh) * rsh);
    o.y = tanhf((x.y - mux) * rsx + (h.y - muh) * rsh);
    o.z = tanhf((x.z - mux) * rsx + (h.z - muh) * rsh);
    o.w = tanhf((x.w - mux) * rsx + (h.w - muh) * rsh);
    ((float4*)(out + (size_t)row * D_DIM))[t] = o;
}

// ---------------- launcher ----------------
extern "C" void kernel_launch(void* const* d_in, const int* in_sizes, int n_in,
                              void* d_out, int out_size, void* d_ws, size_t ws_size,
                              hipStream_t stream) {
    const float* input_ = (const float*)d_in[0];   // [B,1024]
    const float* hidden = (const float*)d_in[1];   // [B,1024]
    const float* Wi     = (const float*)d_in[2];   // [1024,1024]
    const float* bi     = (const float*)d_in[3];   // [1024]
    const float* Wr     = (const float*)d_in[4];   // [1024,1024]
    const float* br     = (const float*)d_in[5];   // [1024]

    const int B = in_sizes[0] / D_DIM;             // 4096

    float*    X      = (float*)d_out;                                    // x-GEMM -> d_out
    float*    H      = (float*)d_ws;                                     // h-GEMM -> ws
    unsigned* scales = (unsigned*)((char*)d_ws + (size_t)B * D_DIM * 4); // 2 uints after H

    k_zero_scales<<<1, 64, 0, stream>>>(scales);
    k_absmax<<<64, 256, 0, stream>>>(Wi, (D_DIM * D_DIM) / 4, scales + 0);
    k_absmax<<<64, 256, 0, stream>>>(Wr, (D_DIM * D_DIM) / 4, scales + 1);

    dim3 grid(D_DIM / TILE_N, B / TILE_M, 1);      // 8 x 32
    k_gemm_qw<<<grid, 256, 0, stream>>>(input_, Wi, bi, scales + 0, X);
    k_gemm_qw<<<grid, 256, 0, stream>>>(hidden, Wr, br, scales + 1, H);

    k_ln_tanh<<<B, 256, 0, stream>>>(X, H, X);
}
